// CodebookGenerator_81192061764457
// MI455X (gfx1250) — matrix operand order repaired and checked by reading, not compiled, not run on hardware
//
#include <hip/hip_runtime.h>
#include <hip/hip_bf16.h>

// ---------------------------------------------------------------------------
// CodebookGenerator (online k-means EMA update) for MI455X / gfx1250.
//
// Pipeline (all on `stream`, serialized by stream order):
//   1) rownorm(dictionary) -> nd bf16   [8192 x 384]
//   2) rownorm(feature)    -> nf bf16   [32768 x 384]
//   3) argmax GEMM: sim = nf @ nd^T via v_wmma_f32_16x16x32_bf16,
//      double-buffered LDS dictionary tiles, fused running argmax
//   4) zero ema/count scratch
//   5) scatter: segment-sum of raw f32 features via global atomics
//   6) finalize: EMA update -> new_dict (f32 output)
// ---------------------------------------------------------------------------

#define NUM_WORDS     8192
#define NUM_CHANNELS  384
#define N_TOKENS      32768
#define NORMALIZE_EPS 1e-5f
#define DIC_MOMENTUM  0.99f

typedef __attribute__((ext_vector_type(16))) __bf16       v16bf;
typedef __attribute__((ext_vector_type(8)))  __bf16       v8bf;
typedef __attribute__((ext_vector_type(8)))  float        v8f;
typedef __attribute__((ext_vector_type(4)))  unsigned int v4u;

// ---------------------------------------------------------------------------
// Row-wise L2 normalize (f32 in) -> bf16 out. One wave32 per row.
// ---------------------------------------------------------------------------
template <int ROWS_TOTAL>
__global__ __launch_bounds__(128)
void rownorm_bf16_kernel(const float* __restrict__ x, __bf16* __restrict__ out) {
  const int wave = threadIdx.x >> 5;   // 4 waves per block, 1 row each
  const int lane = threadIdx.x & 31;
  const int row  = blockIdx.x * 4 + wave;
  if (row >= ROWS_TOTAL) return;

  const float* xr = x + (size_t)row * NUM_CHANNELS;
  float ss = 0.f;
#pragma unroll
  for (int j = 0; j < NUM_CHANNELS / 32; ++j) {
    float v = xr[lane + 32 * j];
    ss += v * v;
  }
#pragma unroll
  for (int m = 16; m >= 1; m >>= 1) ss += __shfl_xor(ss, m, 32);

  const float inv = 1.f / fmaxf(sqrtf(ss), NORMALIZE_EPS);
  __bf16* orow = out + (size_t)row * NUM_CHANNELS;
#pragma unroll
  for (int j = 0; j < NUM_CHANNELS / 32; ++j)
    orow[lane + 32 * j] = (__bf16)(xr[lane + 32 * j] * inv);
}

// ---------------------------------------------------------------------------
// Fused similarity GEMM + argmax, double-buffered.
//   Block: 256 threads = 8 waves; each wave owns a 16-row feature tile with
//   A fragments resident in VGPRs (12 x v16bf = 96 VGPRs) for the whole
//   word loop. Per iteration:
//     - issue b128 global loads of tile wt+1 into registers (latency hidden
//       under the WMMA chain),
//     - run 12 x v_wmma_f32_16x16x32_bf16 against LDS buffer `cur`,
//     - fold the 16x16 sim tile into the running per-row (max, argmax),
//     - b128-store the prefetched tile into LDS buffer `1-cur`,
//     - ONE barrier (separates this store from the previous iteration's
//       reads of that buffer and the next iteration's reads of it).
//
//   D-matrix layout (ISA 7.12.2): lane holds word column (lane&15);
//   acc[i] is feature row i + 8*(lane>>4). Final reduce: shfl_xor over the
//   16 lanes of each half, ties resolved toward the smaller word index
//   (matches jnp.argmax first-occurrence semantics).
// ---------------------------------------------------------------------------
#define WAVES_PER_BLOCK 8
#define ROWS_PER_BLOCK  (WAVES_PER_BLOCK * 16)       // 128
#define KSTEPS          (NUM_CHANNELS / 32)          // 12
#define WORD_TILES      (NUM_WORDS / 16)             // 512
#define TILE_V4         (16 * NUM_CHANNELS * 2 / 16) // 768 x 16B = 12 KB
#define CPY_PER_THREAD  (TILE_V4 / 256)              // 3 x b128 per thread

__global__ __launch_bounds__(256)
void argmax_sim_kernel(const __bf16* __restrict__ nf,
                       const __bf16* __restrict__ nd,
                       int* __restrict__ sim_index) {
  __shared__ __attribute__((aligned(16))) __bf16 sB[2][16 * NUM_CHANNELS];

  const int tid  = threadIdx.x;
  const int wave = tid >> 5;
  const int lane = tid & 31;
  const int half = lane >> 4;   // K-half selector for A/B fragments
  const int m    = lane & 15;   // A: feature row in tile; B/D: word column
  const int row0 = blockIdx.x * ROWS_PER_BLOCK + wave * 16;

  const v4u* nd4 = (const v4u*)nd;

  // Preload this wave's A fragments (16 rows x 384 channels), kept in VGPRs.
  // A 16x32 bf16 layout: half 0 -> K {kk..kk+7, kk+16..kk+23},
  //                      half 1 -> K {kk+8..kk+15, kk+24..kk+31}.
  v16bf afrag[KSTEPS];
  {
    const __bf16* arow = nf + (size_t)(row0 + m) * NUM_CHANNELS;
#pragma unroll
    for (int ks = 0; ks < KSTEPS; ++ks) {
      const int kk = ks * 32;
      v8bf lo = *(const v8bf*)(arow + kk + half * 8);
      v8bf hi = *(const v8bf*)(arow + kk + 16 + half * 8);
      v16bf a;
#pragma unroll
      for (int i = 0; i < 8; ++i) { a[i] = lo[i]; a[i + 8] = hi[i]; }
      afrag[ks] = a;
    }
  }

  float bestv[8];
  int   besti[8];
#pragma unroll
  for (int i = 0; i < 8; ++i) { bestv[i] = -3.4e38f; besti[i] = 0; }

  // Prologue: stage tile 0 into buffer 0 (b128 copies).
  {
    v4u* dst = (v4u*)sB[0];
#pragma unroll
    for (int i = 0; i < CPY_PER_THREAD; ++i)
      dst[tid + 256 * i] = nd4[tid + 256 * i];
  }
  __syncthreads();

  v4u pre[CPY_PER_THREAD];
  for (int wt = 0; wt < WORD_TILES; ++wt) {
    const int cur = wt & 1;

    // Issue next tile's b128 global loads now; the s_wait_loadcnt lands
    // after the WMMA chain, so L2 latency is hidden under compute.
    if (wt + 1 < WORD_TILES) {
      const v4u* src = nd4 + (size_t)(wt + 1) * TILE_V4;
#pragma unroll
      for (int i = 0; i < CPY_PER_THREAD; ++i)
        pre[i] = src[tid + 256 * i];
    }

    // Compute 16x16 sim tile: 12 k-steps of v_wmma_f32_16x16x32_bf16.
    v8f acc = {};
#pragma unroll
    for (int ks = 0; ks < KSTEPS; ++ks) {
      // B 32x16 bf16 layout: lane = column (word m), K = {half*8..+7, 16+half*8..+7}.
      const __bf16* bp = sB[cur] + m * NUM_CHANNELS + ks * 32 + half * 8;
      v8bf lo = *(const v8bf*)(bp);
      v8bf hi = *(const v8bf*)(bp + 16);
      v16bf b;
#pragma unroll
      for (int i = 0; i < 8; ++i) { b[i] = lo[i]; b[i + 8] = hi[i]; }
      acc = __builtin_amdgcn_wmma_f32_16x16x32_bf16(
          /*neg_a=*/false, afrag[ks], /*neg_b=*/false, b,
          /*c_mod=*/(short)0, acc, /*reuse_a=*/false, /*reuse_b=*/false);
    }

    // Fold into running argmax (strict '>' keeps the earliest word on ties).
    const int word = wt * 16 + m;  // this lane's word column
#pragma unroll
    for (int i = 0; i < 8; ++i) {
      if (acc[i] > bestv[i]) { bestv[i] = acc[i]; besti[i] = word; }
    }

    // Store prefetched tile into the other buffer, then ONE barrier.
    if (wt + 1 < WORD_TILES) {
      v4u* dst = (v4u*)sB[1 - cur];
#pragma unroll
      for (int i = 0; i < CPY_PER_THREAD; ++i)
        dst[tid + 256 * i] = pre[i];
    }
    __syncthreads();
  }

  // Reduce (max, argmax) across the 16 lanes of each half.
#pragma unroll
  for (int i = 0; i < 8; ++i) {
    float v  = bestv[i];
    int   id = besti[i];
#pragma unroll
    for (int msk = 1; msk <= 8; msk <<= 1) {
      float ov = __shfl_xor(v, msk, 32);
      int   oi = __shfl_xor(id, msk, 32);
      if (ov > v || (ov == v && oi < id)) { v = ov; id = oi; }
    }
    besti[i] = id;
  }
  if (m == 0) {  // lanes 0 and 16: rows 0..7 and 8..15 of this wave's tile
#pragma unroll
    for (int i = 0; i < 8; ++i)
      sim_index[row0 + half * 8 + i] = besti[i];
  }
}

// ---------------------------------------------------------------------------
// Zero the segment-sum scratch (must run every launch: ws is not re-poisoned).
// ---------------------------------------------------------------------------
__global__ __launch_bounds__(256)
void zero_ws_kernel(float* __restrict__ ema, float* __restrict__ counts) {
  const size_t idx = (size_t)blockIdx.x * blockDim.x + threadIdx.x;
  if (idx < (size_t)NUM_WORDS * NUM_CHANNELS) ema[idx] = 0.f;
  if (idx < NUM_WORDS) counts[idx] = 0.f;
}

// ---------------------------------------------------------------------------
// Segment-sum scatter: one block per token, global f32 atomics into [K,C].
// ---------------------------------------------------------------------------
__global__ __launch_bounds__(128)
void scatter_kernel(const float* __restrict__ feature,
                    const int* __restrict__ sim_index,
                    float* __restrict__ ema, float* __restrict__ counts) {
  const int token = blockIdx.x;
  const int idx   = sim_index[token];
  const float* fr = feature + (size_t)token * NUM_CHANNELS;
  float* er       = ema + (size_t)idx * NUM_CHANNELS;
  for (int c = threadIdx.x; c < NUM_CHANNELS; c += blockDim.x)
    atomicAdd(er + c, fr[c]);
  if (threadIdx.x == 0) atomicAdd(counts + idx, 1.0f);
}

// ---------------------------------------------------------------------------
// EMA update: new_dict = used ? (m*sum + (1-m)*ema) / (m*num + (1-m)*cnt)
//                             : dictionary
// ---------------------------------------------------------------------------
__global__ __launch_bounds__(128)
void finalize_kernel(const float* __restrict__ dict,
                     const float* __restrict__ dsum,
                     const float* __restrict__ dnum,
                     const float* __restrict__ ema,
                     const float* __restrict__ counts,
                     float* __restrict__ out) {
  const int w = blockIdx.x;
  const float cnt    = counts[w];
  const bool  used   = cnt > 0.f;
  const float mm     = DIC_MOMENTUM;
  const float newnum = mm * dnum[w] + (1.f - mm) * cnt;
  for (int c = threadIdx.x; c < NUM_CHANNELS; c += blockDim.x) {
    const size_t o  = (size_t)w * NUM_CHANNELS + c;
    const float  ns = mm * dsum[o] + (1.f - mm) * ema[o];
    out[o] = used ? (ns / newnum) : dict[o];
  }
}

// ---------------------------------------------------------------------------
// Launcher
// ---------------------------------------------------------------------------
extern "C" void kernel_launch(void* const* d_in, const int* in_sizes, int n_in,
                              void* d_out, int out_size, void* d_ws, size_t ws_size,
                              hipStream_t stream) {
  const float* feature = (const float*)d_in[0];  // [32768, 384]
  const float* dict    = (const float*)d_in[1];  // [8192, 384]
  const float* dsum    = (const float*)d_in[2];  // [8192, 384]
  const float* dnum    = (const float*)d_in[3];  // [8192]
  float*       out     = (float*)d_out;          // [8192, 384]

  // Workspace layout (all 256B-aligned segments), total ~44.2 MB:
  char* ws = (char*)d_ws;
  __bf16* nf     = (__bf16*)(ws);                            // 25,165,824 B
  __bf16* nd     = (__bf16*)(ws + 25165824);                 //  6,291,456 B
  int*    simidx = (int*)   (ws + 25165824 + 6291456);       //    131,072 B
  float*  ema    = (float*) (ws + 25165824 + 6291456 + 131072);      // 12,582,912 B
  float*  counts = (float*) (ws + 25165824 + 6291456 + 131072 + 12582912); // 32,768 B

  // 1-2) L2 normalize + bf16 round (one wave per row, 4 rows per block)
  rownorm_bf16_kernel<NUM_WORDS><<<NUM_WORDS / 4, 128, 0, stream>>>(dict, nd);
  rownorm_bf16_kernel<N_TOKENS><<<N_TOKENS / 4, 128, 0, stream>>>(feature, nf);

  // 3) WMMA similarity + fused argmax: 128 rows per 256-thread block
  argmax_sim_kernel<<<N_TOKENS / ROWS_PER_BLOCK, 256, 0, stream>>>(nf, nd, simidx);

  // 4) zero scratch, 5) segment-sum scatter, 6) EMA finalize
  zero_ws_kernel<<<((size_t)NUM_WORDS * NUM_CHANNELS + 255) / 256, 256, 0, stream>>>(ema, counts);
  scatter_kernel<<<N_TOKENS, 128, 0, stream>>>(feature, simidx, ema, counts);
  finalize_kernel<<<NUM_WORDS, 128, 0, stream>>>(dict, dsum, dnum, ema, counts, out);
}
